// Model_2980707303634
// MI455X (gfx1250) — matrix operand order repaired and checked by reading, not compiled, not run on hardware
//
#include <hip/hip_runtime.h>

typedef __bf16 bf16_t;
typedef __attribute__((ext_vector_type(16))) __bf16 v16bf;
typedef __attribute__((ext_vector_type(8)))  float  v8f;
typedef unsigned int u32x4 __attribute__((ext_vector_type(4)));
typedef int          i32x8 __attribute__((ext_vector_type(8)));
typedef int          i32x4 __attribute__((ext_vector_type(4)));

#define S_LEN   2048
#define H_DIM   1024
#define HD      64
#define NH      16
#define DEPTH   8
#define VOCAB   256
#define BATCH   4
#define FF_DIM  4096
#define NTOK    (BATCH * S_LEN)   // 8192

#if defined(__has_builtin)
#if __has_builtin(__builtin_amdgcn_tensor_load_to_lds)
#define HAVE_TDM 1
#endif
#endif
#ifndef HAVE_TDM
#define HAVE_TDM 0
#endif

// Fragment <-> 16B-chunk bit-cast helper (two ds_load_b128 per fragment).
union Frag16 {
  uint4 u[2];
  v16bf v;
};

// LDS byte offset (generic LDS addresses carry the offset in the low 32 bits).
__device__ __forceinline__ unsigned lds_off(const void* p) {
  return (unsigned)(unsigned long long)p;
}

// Async 16B global -> LDS copy (ASYNCcnt-tracked, ISA 08 section 4).
__device__ __forceinline__ void async_cp16(unsigned lds, const void* gptr) {
  unsigned long long ga = (unsigned long long)gptr;
  asm volatile("global_load_async_to_lds_b128 %0, %1, off"
               :: "v"(lds), "v"(ga) : "memory");
}

__device__ __forceinline__ void async_wait0() {
  asm volatile("s_wait_asynccnt 0x0" ::: "memory");
}

#if HAVE_TDM
// TDM: DMA one 64x64 bf16 tile (row stride = strideElems) into LDS.
// D# packing per ISA 08 section 8: group0 = {flags, lds_addr, global_addr,
// type=2}; group1 = {data_size=2B, tensor dims, tile dims, dim0 stride}.
__device__ __forceinline__ void tdm_load_tile64(unsigned ldsOff, const void* g,
                                                unsigned strideElems) {
  unsigned long long ga = (unsigned long long)g;
  const unsigned dim0 = 0x400000u, dim1 = 0x400000u;  // large (no OOB clip)
  u32x4 g0;
  g0[0] = 1u;                                           // count=1, user mode
  g0[1] = ldsOff;                                       // lds_addr
  g0[2] = (unsigned)ga;                                 // global_addr[31:0]
  g0[3] = (unsigned)((ga >> 32) & 0x01FFFFFFull) | 0x80000000u;  // [56:32]|type=2
  i32x8 g1;
  g1[0] = 0x00010000;                                   // data_size=1 (2 bytes)
  g1[1] = (int)((dim0 & 0xffffu) << 16);                // tensor_dim0 lo16
  g1[2] = (int)((dim0 >> 16) | ((dim1 & 0xffffu) << 16));
  g1[3] = (int)((dim1 >> 16) | (64u << 16));            // tile_dim0 = 64
  g1[4] = (int)64;                                      // tile_dim1 = 64
  g1[5] = (int)strideElems;                             // dim0 stride lo32
  g1[6] = 0;                                            // stride hi16
  g1[7] = 0;
  i32x4 gz = {0, 0, 0, 0};
#if __clang_major__ >= 23
  i32x8 gz8 = {0, 0, 0, 0, 0, 0, 0, 0};
  __builtin_amdgcn_tensor_load_to_lds(g0, g1, gz, gz, gz8, 0);
#else
  __builtin_amdgcn_tensor_load_to_lds(g0, g1, gz, gz, 0);
#endif
}
#endif

// ---------------------------------------------------------------------------
// fp32 -> bf16 transposing conversion: in [Kd][Nd] fp32 -> out [Nd][Kd] bf16.
// ---------------------------------------------------------------------------
__global__ void f32_to_bf16_T_k(const float* __restrict__ in,
                                bf16_t* __restrict__ out, int Kd, int Nd) {
  int idx = blockIdx.x * 256 + threadIdx.x;
  if (idx < Kd * Nd) {
    int n = idx / Kd, k = idx % Kd;
    out[idx] = (bf16_t)in[(size_t)k * Nd + n];
  }
}

// ---------------------------------------------------------------------------
// Embedding + positional add
// ---------------------------------------------------------------------------
__global__ __launch_bounds__(256) void embed_k(const int* __restrict__ idx,
                                               const float* __restrict__ emb,
                                               const float* __restrict__ pos,
                                               float* __restrict__ x) {
  const int bs = blockIdx.x;
  const int b = bs / S_LEN, s = bs % S_LEN;
  const int tok = idx[b * (S_LEN + 1) + s];
  const int tid = threadIdx.x;
#pragma unroll
  for (int i = 0; i < 4; ++i) {
    int h = tid * 4 + i;
    x[(size_t)bs * H_DIM + h] =
        emb[(size_t)tok * H_DIM + h] + pos[(size_t)s * H_DIM + h];
  }
}

// ---------------------------------------------------------------------------
// LayerNorm over H=1024 (one row per block; fp32 and/or bf16 out)
// ---------------------------------------------------------------------------
__global__ __launch_bounds__(256) void layernorm_k(
    const float* __restrict__ x, const float* __restrict__ g,
    const float* __restrict__ bta, float* __restrict__ outF,
    bf16_t* __restrict__ outB) {
  __shared__ float red[256];
  const int row = blockIdx.x;
  const int tid = threadIdx.x;
  const float* xr = x + (size_t)row * H_DIM;

  float v[4];
  float s = 0.0f;
#pragma unroll
  for (int i = 0; i < 4; ++i) { v[i] = xr[tid * 4 + i]; s += v[i]; }
  red[tid] = s;
  __syncthreads();
  for (int st = 128; st > 0; st >>= 1) {
    if (tid < st) red[tid] += red[tid + st];
    __syncthreads();
  }
  const float mu = red[0] * (1.0f / H_DIM);
  __syncthreads();

  float vs = 0.0f;
#pragma unroll
  for (int i = 0; i < 4; ++i) { float d = v[i] - mu; vs += d * d; }
  red[tid] = vs;
  __syncthreads();
  for (int st = 128; st > 0; st >>= 1) {
    if (tid < st) red[tid] += red[tid + st];
    __syncthreads();
  }
  const float rstd = rsqrtf(red[0] * (1.0f / H_DIM) + 1e-5f);

#pragma unroll
  for (int i = 0; i < 4; ++i) {
    int c = tid * 4 + i;
    float o = (v[i] - mu) * rstd * g[c] + bta[c];
    if (outF) outF[(size_t)row * H_DIM + c] = o;
    if (outB) outB[(size_t)row * H_DIM + c] = (bf16_t)o;
  }
}

// ---------------------------------------------------------------------------
// bf16 WMMA GEMM: C(MxN) = A(MxK) * BwT^T + bias (BwT is [N][K] transposed).
// Block tile 64x64, K-step 64, DOUBLE-BUFFERED LDS: the async engine (A tile,
// per-lane b128 copies) and TDM (B tile) fill buf^1 while WMMA consumes buf.
//   EPI 0: bf16 out | 1: fp32 residual add | 2: exact GELU->bf16
//   EPI 3: fp32 out | 4: q/k -> outB [tok][2H], v -> outB2 [b,h,d,s]
// ---------------------------------------------------------------------------
template <int EPI>
__global__ __launch_bounds__(256) void gemm_bf16_k(
    const bf16_t* __restrict__ A, const bf16_t* __restrict__ BwT,
    const float* __restrict__ bias, const float* __restrict__ resid,
    float* __restrict__ outF, bf16_t* __restrict__ outB,
    bf16_t* __restrict__ outB2, int M, int N, int K) {
  __shared__ __align__(16) bf16_t As[2][64][64];   // [buf][m][k]
  __shared__ __align__(16) bf16_t Bt[2][64][64];   // [buf][n][k]

  const int tid = threadIdx.x;
  const int lane = tid & 31;
  const int w = tid >> 5;
  const int wm = w >> 1;      // 0..3
  const int wn = w & 1;       // 0..1
  const int mBlk = blockIdx.y * 64;
  const int nBlk = blockIdx.x * 64;
  const int l15 = lane & 15;
  const int kA = (lane < 16) ? 0 : 8;    // A-frag K base (ISA 7.12.2)
  const int kB = (lane < 16) ? 0 : 8;    // B-frag K base

  const int tRow = tid >> 2;             // 0..63
  const int tCol = (tid & 3) * 16;       // 0,16,32,48

  auto stage = [&](int k0, int buf) {
    const bf16_t* ap = &A[(size_t)(mBlk + tRow) * K + k0 + tCol];
    async_cp16(lds_off(&As[buf][tRow][tCol]), ap);
    async_cp16(lds_off(&As[buf][tRow][tCol + 8]), ap + 8);
#if HAVE_TDM
    if (w == 0)
      tdm_load_tile64(lds_off(&Bt[buf][0][0]),
                      &BwT[(size_t)nBlk * K + k0], (unsigned)K);
#else
    const bf16_t* bp = &BwT[(size_t)(nBlk + tRow) * K + k0 + tCol];
    async_cp16(lds_off(&Bt[buf][tRow][tCol]), bp);
    async_cp16(lds_off(&Bt[buf][tRow][tCol + 8]), bp + 8);
#endif
  };

  v8f acc[2] = {};

  const int nIter = K >> 6;
  stage(0, 0);
  for (int i = 0; i < nIter; ++i) {
    const int buf = i & 1;
    async_wait0();
#if HAVE_TDM
    if (w == 0) __builtin_amdgcn_s_wait_tensorcnt(0);
#endif
    __syncthreads();
    // overlap: fill the other buffer while this one is consumed
    if (i + 1 < nIter) stage((i + 1) << 6, buf ^ 1);

#pragma unroll
    for (int t = 0; t < 2; ++t) {
      Frag16 af;
      af.u[0] = *reinterpret_cast<const uint4*>(
          &As[buf][wm * 16 + l15][t * 32 + kA]);
      af.u[1] = *reinterpret_cast<const uint4*>(
          &As[buf][wm * 16 + l15][t * 32 + kA + 16]);
#pragma unroll
      for (int nn = 0; nn < 2; ++nn) {
        Frag16 bfr;
        const bf16_t* br = &Bt[buf][wn * 32 + nn * 16 + l15][t * 32 + kB];
        bfr.u[0] = *reinterpret_cast<const uint4*>(br);
        bfr.u[1] = *reinterpret_cast<const uint4*>(br + 16);
        acc[nn] = __builtin_amdgcn_wmma_f32_16x16x32_bf16(
            false, af.v, false, bfr.v, (short)0, acc[nn], false, false);
      }
    }
  }

#pragma unroll
  for (int nn = 0; nn < 2; ++nn) {
#pragma unroll
    for (int r = 0; r < 8; ++r) {
      const int row = mBlk + wm * 16 + r + ((lane < 16) ? 0 : 8);
      const int col = nBlk + wn * 32 + nn * 16 + l15;
      float val = acc[nn][r] + bias[col];
      if (EPI == 0) {
        outB[(size_t)row * N + col] = (bf16_t)val;
      } else if (EPI == 1) {
        const size_t o = (size_t)row * N + col;
        outF[o] = resid[o] + val;
      } else if (EPI == 2) {
        float gg = 0.5f * val * (1.0f + erff(val * 0.70710678118654752f));
        outB[(size_t)row * N + col] = (bf16_t)gg;
      } else if (EPI == 3) {
        outF[(size_t)row * N + col] = val;
      } else {  // EPI 4: split QKV
        if (col < 2 * H_DIM) {
          outB[(size_t)row * (2 * H_DIM) + col] = (bf16_t)val;
        } else {
          const int f = col - 2 * H_DIM;
          const int hd = f >> 6, d = f & 63;
          const int bb = row >> 11, ss = row & (S_LEN - 1);
          outB2[(((size_t)bb * NH + hd) * HD + d) * S_LEN + ss] = (bf16_t)val;
        }
      }
    }
  }
}

// ---------------------------------------------------------------------------
// Causal flash attention, WMMA for Q*K^T and P*V, 64-key tiles,
// double-buffered async K/V staging overlapping softmax + WMMA.
// Grid (S/64, NH, B); block 128 = 4 waves; each wave owns 16 query rows.
// qk: [tok][2H] bf16 (q cols 0..1023, k cols 1024..2047);
// vT: [b,h,d,s] bf16 (so the V tile is a pass-through async copy).
// ---------------------------------------------------------------------------
__global__ __launch_bounds__(128) void attn_k(const bf16_t* __restrict__ qk,
                                              const bf16_t* __restrict__ vT,
                                              bf16_t* __restrict__ outB) {
  __shared__ __align__(16) bf16_t Ks[2][64][64];   // [buf][key][d]
  __shared__ __align__(16) bf16_t Vt[2][64][64];   // [buf][d][key]
  __shared__ __align__(16) bf16_t Pl[4][16][64];   // per-wave P tile

  const int tid = threadIdx.x;
  const int lane = tid & 31;
  const int w = tid >> 5;
  const int l15 = lane & 15;
  const int kA = (lane < 16) ? 0 : 8;
  const int kB = (lane < 16) ? 0 : 8;
  const int halfOff = (lane < 16) ? 0 : 8;
  const int head = blockIdx.y;
  const int b = blockIdx.z;
  const int q0b = blockIdx.x * 64;
  const int q0 = q0b + w * 16;

  const size_t tokStride = 2 * H_DIM;
  const bf16_t* Q  = qk + (size_t)b * S_LEN * tokStride + head * HD;
  const bf16_t* Kp = Q + H_DIM;
  const bf16_t* vTh = vT + ((size_t)b * NH + head) * HD * S_LEN;

  // Q A-fragments over d = 0..31 and 32..63 (b128 global loads)
  Frag16 qa[2];
#pragma unroll
  for (int t = 0; t < 2; ++t) {
    const bf16_t* qp = &Q[(size_t)(q0 + l15) * tokStride + t * 32 + kA];
    qa[t].u[0] = *reinterpret_cast<const uint4*>(qp);
    qa[t].u[1] = *reinterpret_cast<const uint4*>(qp + 16);
  }

  // staging geometry: 64 rows x 64 cols per tile; 128 threads -> half row each
  const int sRow = tid >> 1;             // 0..63
  const int sCol = (tid & 1) * 32;       // 0,32

  auto stageKV = [&](int kb, int buf) {
    const bf16_t* kp = &Kp[(size_t)(kb + sRow) * tokStride + sCol];
    const bf16_t* vp = &vTh[(size_t)sRow * S_LEN + kb + sCol];
#pragma unroll
    for (int c = 0; c < 4; ++c) {
      async_cp16(lds_off(&Ks[buf][sRow][sCol + c * 8]), kp + c * 8);
      async_cp16(lds_off(&Vt[buf][sRow][sCol + c * 8]), vp + c * 8);
    }
  };

  v8f zero = {};
  v8f accd[4];
#pragma unroll
  for (int i = 0; i < 4; ++i) accd[i] = zero;

  float mrow[8], lrow[8];
#pragma unroll
  for (int r = 0; r < 8; ++r) { mrow[r] = -1e30f; lrow[r] = 0.0f; }

  // 64-aligned q block + 64-wide key tiles: every wave needs the same tiles.
  const int nkt = (q0b >> 6) + 1;

  stageKV(0, 0);
  for (int kt = 0; kt < nkt; ++kt) {
    const int kb = kt * 64;
    const int buf = kt & 1;
    async_wait0();
    __syncthreads();
    if (kt + 1 < nkt) stageKV((kt + 1) * 64, buf ^ 1);

    // scores 16x64: four 16x16 WMMA column tiles, K over d (2x32)
    v8f sc[4];
#pragma unroll
    for (int ct = 0; ct < 4; ++ct) {
      v8f s = zero;
#pragma unroll
      for (int t = 0; t < 2; ++t) {
        Frag16 kf;
        const bf16_t* kr = &Ks[buf][ct * 16 + l15][t * 32 + kB];
        kf.u[0] = *reinterpret_cast<const uint4*>(kr);
        kf.u[1] = *reinterpret_cast<const uint4*>(kr + 16);
        s = __builtin_amdgcn_wmma_f32_16x16x32_bf16(
            false, qa[t].v, false, kf.v, (short)0, s, false, false);
      }
      sc[ct] = s;
    }

    // online softmax (row stats lane-uniform within each 16-lane half)
#pragma unroll
    for (int r = 0; r < 8; ++r) {
      const int row = q0 + r + halfOff;
      float sv[4], rm = -1e30f;
#pragma unroll
      for (int ct = 0; ct < 4; ++ct) {
        const int c = kb + ct * 16 + l15;
        sv[ct] = sc[ct][r] * 0.125f + ((c > row) ? -10000.0f : 0.0f);
        rm = fmaxf(rm, sv[ct]);
      }
#pragma unroll
      for (int d = 1; d < 16; d <<= 1) rm = fmaxf(rm, __shfl_xor(rm, d, 32));
      const float mn = fmaxf(mrow[r], rm);
      const float alpha = __expf(mrow[r] - mn);
      float rs = 0.0f;
#pragma unroll
      for (int ct = 0; ct < 4; ++ct) {
        const float p = __expf(sv[ct] - mn);
        rs += p;
        Pl[w][r + halfOff][ct * 16 + l15] = (bf16_t)p;
      }
#pragma unroll
      for (int d = 1; d < 16; d <<= 1) rs += __shfl_xor(rs, d, 32);
      lrow[r] = lrow[r] * alpha + rs;
      mrow[r] = mn;
#pragma unroll
      for (int ct = 0; ct < 4; ++ct) accd[ct][r] *= alpha;
    }

    // repack P (C layout -> A layout, keys 0..31 / 32..63) via per-wave LDS
    Frag16 pa[2];
#pragma unroll
    for (int h = 0; h < 2; ++h) {
      pa[h].u[0] = *reinterpret_cast<const uint4*>(&Pl[w][l15][h * 32 + kA]);
      pa[h].u[1] =
          *reinterpret_cast<const uint4*>(&Pl[w][l15][h * 32 + kA + 16]);
    }

    // out(16x64) += P(16x64) * V(64x64): 4 d-tiles x 2 key halves
#pragma unroll
    for (int ct = 0; ct < 4; ++ct) {
#pragma unroll
      for (int h = 0; h < 2; ++h) {
        Frag16 vf;
        const bf16_t* vrp = &Vt[buf][ct * 16 + l15][h * 32 + kB];
        vf.u[0] = *reinterpret_cast<const uint4*>(vrp);
        vf.u[1] = *reinterpret_cast<const uint4*>(vrp + 16);
        accd[ct] = __builtin_amdgcn_wmma_f32_16x16x32_bf16(
            false, pa[h].v, false, vf.v, (short)0, accd[ct], false, false);
      }
    }
  }

#pragma unroll
  for (int r = 0; r < 8; ++r) {
    const int row = q0 + r + halfOff;
    const float inv = 1.0f / lrow[r];
#pragma unroll
    for (int ct = 0; ct < 4; ++ct) {
      const int dcol = ct * 16 + l15;
      outB[((size_t)b * S_LEN + row) * H_DIM + head * HD + dcol] =
          (bf16_t)(accd[ct][r] * inv);
    }
  }
}

// ---------------------------------------------------------------------------
// log-softmax + NLL gather + mean (one wave per 256-logit row)
// ---------------------------------------------------------------------------
__global__ void zero_k(float* p) {
  if (threadIdx.x == 0 && blockIdx.x == 0) p[0] = 0.0f;
}

__global__ __launch_bounds__(256) void loss_k(const float* __restrict__ logits,
                                              const int* __restrict__ idx,
                                              float* __restrict__ out) {
  const int lane = threadIdx.x & 31;
  const int w = threadIdx.x >> 5;
  const int row = blockIdx.x * 8 + w;
  const int b = row / S_LEN, s = row % S_LEN;
  const int tgt = idx[b * (S_LEN + 1) + s + 1];
  const float* lr = logits + (size_t)row * VOCAB;

  float lv[8];
  float m = -1e30f;
#pragma unroll
  for (int j = 0; j < 8; ++j) {
    lv[j] = lr[lane + 32 * j];
    m = fmaxf(m, lv[j]);
  }
#pragma unroll
  for (int d = 1; d < 32; d <<= 1) m = fmaxf(m, __shfl_xor(m, d, 32));
  float sum = 0.0f;
#pragma unroll
  for (int j = 0; j < 8; ++j) sum += __expf(lv[j] - m);
#pragma unroll
  for (int d = 1; d < 32; d <<= 1) sum += __shfl_xor(sum, d, 32);
  if (lane == 0) {
    const float lpt = lr[tgt] - m - __logf(sum);
    atomicAdd(out, -lpt * (1.0f / NTOK));
  }
}

// ---------------------------------------------------------------------------
// Launch. Input order: 0 indices | 1 embed | 2 pos | 3 ln0_g | 4 ln0_b
// per layer l (base=5+12l): ln1_g ln1_b Wqkv bqkv Wproj bproj ln2_g ln2_b
//                           Wup bup Wdown bdown
// 101 lnf_g | 102 lnf_b | 103 Wout | 104 bout
// ---------------------------------------------------------------------------
extern "C" void kernel_launch(void* const* d_in, const int* in_sizes, int n_in,
                              void* d_out, int out_size, void* d_ws,
                              size_t ws_size, hipStream_t stream) {
  (void)in_sizes; (void)n_in; (void)out_size; (void)ws_size;
  const int* indices = (const int*)d_in[0];
  const float* emb   = (const float*)d_in[1];
  const float* pos   = (const float*)d_in[2];
  const float* ln0_g = (const float*)d_in[3];
  const float* ln0_b = (const float*)d_in[4];
  const float* lnf_g = (const float*)d_in[101];
  const float* lnf_b = (const float*)d_in[102];
  const float* Wout  = (const float*)d_in[103];
  const float* bout  = (const float*)d_in[104];

  char* ws = (char*)d_ws;
  size_t off = 0;
  auto alloc = [&](size_t bytes) -> void* {
    void* p = ws + off;
    off += (bytes + 255) & ~(size_t)255;
    return p;
  };

  float*  x      = (float*)alloc((size_t)NTOK * H_DIM * 4);
  bf16_t* xn     = (bf16_t*)alloc((size_t)NTOK * H_DIM * 2);
  bf16_t* qkb    = (bf16_t*)alloc((size_t)NTOK * 2 * H_DIM * 2);
  bf16_t* vTb    = (bf16_t*)alloc((size_t)NTOK * H_DIM * 2);
  bf16_t* attnb  = (bf16_t*)alloc((size_t)NTOK * H_DIM * 2);
  bf16_t* ffh    = (bf16_t*)alloc((size_t)NTOK * FF_DIM * 2);
  float*  logits = (float*)alloc((size_t)NTOK * VOCAB * 4);

  bf16_t* wqkv_t[DEPTH];
  bf16_t* wproj_t[DEPTH];
  bf16_t* wup_t[DEPTH];
  bf16_t* wdown_t[DEPTH];
  for (int l = 0; l < DEPTH; ++l) {
    wqkv_t[l]  = (bf16_t*)alloc((size_t)H_DIM * 3 * H_DIM * 2);
    wproj_t[l] = (bf16_t*)alloc((size_t)H_DIM * H_DIM * 2);
    wup_t[l]   = (bf16_t*)alloc((size_t)H_DIM * FF_DIM * 2);
    wdown_t[l] = (bf16_t*)alloc((size_t)FF_DIM * H_DIM * 2);
  }
  bf16_t* wout_t = (bf16_t*)alloc((size_t)H_DIM * VOCAB * 2);

  // weights converted AND transposed to [N][K] once per launch
  auto convT = [&](const float* src, bf16_t* dst, int Kd, int Nd) {
    int n = Kd * Nd;
    f32_to_bf16_T_k<<<(n + 255) / 256, 256, 0, stream>>>(src, dst, Kd, Nd);
  };
  for (int l = 0; l < DEPTH; ++l) {
    const int base = 5 + l * 12;
    convT((const float*)d_in[base + 2],  wqkv_t[l],  H_DIM, 3 * H_DIM);
    convT((const float*)d_in[base + 4],  wproj_t[l], H_DIM, H_DIM);
    convT((const float*)d_in[base + 8],  wup_t[l],   H_DIM, FF_DIM);
    convT((const float*)d_in[base + 10], wdown_t[l], FF_DIM, H_DIM);
  }
  convT(Wout, wout_t, H_DIM, VOCAB);

  embed_k<<<NTOK, 256, 0, stream>>>(indices, emb, pos, x);
  layernorm_k<<<NTOK, 256, 0, stream>>>(x, ln0_g, ln0_b, x, nullptr);

  for (int l = 0; l < DEPTH; ++l) {
    const int base = 5 + l * 12;
    const float* ln1_g = (const float*)d_in[base + 0];
    const float* ln1_b = (const float*)d_in[base + 1];
    const float* bqkv  = (const float*)d_in[base + 3];
    const float* bproj = (const float*)d_in[base + 5];
    const float* ln2_g = (const float*)d_in[base + 6];
    const float* ln2_b = (const float*)d_in[base + 7];
    const float* bup   = (const float*)d_in[base + 9];
    const float* bdown = (const float*)d_in[base + 11];

    layernorm_k<<<NTOK, 256, 0, stream>>>(x, ln1_g, ln1_b, nullptr, xn);
    gemm_bf16_k<4><<<dim3(3 * H_DIM / 64, NTOK / 64), 256, 0, stream>>>(
        xn, wqkv_t[l], bqkv, nullptr, nullptr, qkb, vTb,
        NTOK, 3 * H_DIM, H_DIM);
    attn_k<<<dim3(S_LEN / 64, NH, BATCH), 128, 0, stream>>>(qkb, vTb, attnb);
    gemm_bf16_k<1><<<dim3(H_DIM / 64, NTOK / 64), 256, 0, stream>>>(
        attnb, wproj_t[l], bproj, x, x, nullptr, nullptr,
        NTOK, H_DIM, H_DIM);
    layernorm_k<<<NTOK, 256, 0, stream>>>(x, ln2_g, ln2_b, nullptr, xn);
    gemm_bf16_k<2><<<dim3(FF_DIM / 64, NTOK / 64), 256, 0, stream>>>(
        xn, wup_t[l], bup, nullptr, nullptr, ffh, nullptr,
        NTOK, FF_DIM, H_DIM);
    gemm_bf16_k<1><<<dim3(H_DIM / 64, NTOK / 64), 256, 0, stream>>>(
        ffh, wdown_t[l], bdown, x, x, nullptr, nullptr,
        NTOK, H_DIM, FF_DIM);
  }

  layernorm_k<<<NTOK, 256, 0, stream>>>(x, lnf_g, lnf_b, nullptr, xn);
  gemm_bf16_k<3><<<dim3(VOCAB / 64, NTOK / 64), 256, 0, stream>>>(
      xn, wout_t, bout, nullptr, logits, nullptr, nullptr,
      NTOK, VOCAB, H_DIM);

  zero_k<<<1, 64, 0, stream>>>((float*)d_out);
  loss_k<<<NTOK / 8, 256, 0, stream>>>(logits, indices, (float*)d_out);
}
